// RecurrentMNIST_70300024700958
// MI455X (gfx1250) — matrix operand order repaired and checked by reading, not compiled, not run on hardware
//
#include <hip/hip_runtime.h>
#include <cmath>

typedef _Float16 v16h __attribute__((ext_vector_type(16)));
typedef _Float16 v8h  __attribute__((ext_vector_type(8)));
typedef float    v8f  __attribute__((ext_vector_type(8)));
typedef float    v4f  __attribute__((ext_vector_type(4)));

#define B_SZ   2048
#define T_SZ   784
#define H_SZ   100
#define OUT_SZ 10
#define KP     128   // hidden dim padded to 4 x K32 WMMA chunks
#define NW     7     // waves per block: 7 x 16 = 112 >= H

__device__ __forceinline__ float fast_tanh(float x) {
#if __has_builtin(__builtin_amdgcn_tanhf)
    return __builtin_amdgcn_tanhf(x);   // native V_TANH_F32 (gfx1250 transcendental)
#else
    return tanhf(x);
#endif
}

// One workgroup owns 16 batch rows and runs the full 784-step recurrence
// with curr resident in LDS (f16, ping-pong) and its Ws^T slice resident in
// registers in WMMA B-layout. wave w computes hidden outputs n = 16w..16w+15.
__global__ __launch_bounds__(NW * 32)
void RecurrentMNIST_scan_kernel(const float* __restrict__ x,
                                const int*   __restrict__ order,
                                const float* __restrict__ Wi,
                                const float* __restrict__ Ws,
                                const float* __restrict__ bs,
                                const float* __restrict__ Wo,
                                const float* __restrict__ bo,
                                float* __restrict__ out)
{
    __shared__ alignas(16) float    xT[T_SZ * 16];       // x tile, transposed: [t][m]
    __shared__ alignas(16) _Float16 cur[2][16 * KP];     // curr ping-pong: [m][k]

    const int  tid  = threadIdx.x;
    const int  wave = tid >> 5;
    const int  lane = tid & 31;
    const int  nloc = lane & 15;
    const bool hi   = lane >= 16;
    const int  n    = wave * 16 + nloc;     // hidden index this lane produces (0..111)
    const int  mb   = blockIdx.x * 16;      // batch row base

    // ---- Ws^T slice -> registers, WMMA B-matrix layout (loaded once) ----
    // B[k][n] = Ws[n][k]. 16-bit B chunk c: VGPR halves j: lanes0-15 hold
    // k = 32c+2j(+8 if j>=4),+1 ; lanes16-31 hold the same +8 ; n = lane%16 + 16*wave.
    // Pad lanes (n >= 100) get all-zero B columns, zero bias and zero input
    // weight, so their accumulator is exactly 0 every step -> no per-step guard.
    v16h bw[4];
    #pragma unroll
    for (int c = 0; c < 4; ++c) {
        #pragma unroll
        for (int j = 0; j < 8; ++j) {
            int k = 32 * c + 2 * j + (hi ? 8 : 0) + (j >= 4 ? 8 : 0);
            float f0 = (n < H_SZ && k     < H_SZ) ? Ws[n * H_SZ + k]     : 0.f;
            float f1 = (n < H_SZ && k + 1 < H_SZ) ? Ws[n * H_SZ + k + 1] : 0.f;
            bw[c][2 * j]     = (_Float16)f0;
            bw[c][2 * j + 1] = (_Float16)f1;
        }
    }
    const float wiv = (n < H_SZ) ? Wi[n] : 0.f;   // Wi is (H,1)
    const float bsv = (n < H_SZ) ? bs[n] : 0.f;

    // ---- stage x (permuted by `order`, transposed) into LDS; zero curr bufs ----
    for (int i = tid; i < T_SZ * 16; i += NW * 32) {
        int t = i >> 4, m = i & 15;
        int col = order[t];
        xT[t * 16 + m] = x[(size_t)(mb + m) * T_SZ + col];
    }
    for (int i = tid; i < 2 * 16 * KP; i += NW * 32)
        ((_Float16*)cur)[i] = (_Float16)0.f;      // zeros incl. k-pad 112..127
    __syncthreads();

    // ---- recurrence ----
    const int m_a  = lane & 15;      // A row this lane loads
    const int kofs = hi ? 8 : 0;     // A-layout: hi lanes start at k%32 = 8
    int p = 0;
    for (int t = 0; t < T_SZ; ++t) {
        // Issue ALL LDS loads for this step up front so they pipeline:
        // 8x ds_load_b128 for A (4 K-chunks) + 2x ds_load_b128 for x_t.
        const _Float16* cb   = &cur[p][0];
        const _Float16* rowp = cb + m_a * KP + kofs;
        v16h a[4];
        #pragma unroll
        for (int ck = 0; ck < 4; ++ck) {
            v8h alo = *(const v8h*)(rowp + ck * 32);        // k = 32ck+kofs .. +7
            v8h ahi = *(const v8h*)(rowp + ck * 32 + 16);   // k = 32ck+kofs+16 .. +23
            a[ck] = __builtin_shufflevector(alo, ahi,
                        0,1,2,3,4,5,6,7,8,9,10,11,12,13,14,15);
        }
        v4f xa = *(const v4f*)&xT[t * 16 + (hi ? 8 : 0)];
        v4f xb = *(const v4f*)&xT[t * 16 + (hi ? 8 : 0) + 4];

        // C init: bias + per-row input injection (C vgpr v -> m = v (+8 hi))
        v8f c;
        c[0] = bsv + xa[0] * wiv; c[1] = bsv + xa[1] * wiv;
        c[2] = bsv + xa[2] * wiv; c[3] = bsv + xa[3] * wiv;
        c[4] = bsv + xb[0] * wiv; c[5] = bsv + xb[1] * wiv;
        c[6] = bsv + xb[2] * wiv; c[7] = bsv + xb[3] * wiv;

        // K-accumulation chain: 4 back-to-back WMMAs
        #pragma unroll
        for (int ck = 0; ck < 4; ++ck)
            c = __builtin_amdgcn_wmma_f32_16x16x32_f16(
                    false, a[ck], false, bw[ck], (short)0, c, false, false);

        // tanh (native V_TANH_F32) + write new curr (f16).
        // Pad lanes store tanh(0) == 0, keeping pad columns zero by construction.
        _Float16* nb = &cur[p ^ 1][0];
        #pragma unroll
        for (int v = 0; v < 8; ++v) {
            int m = v + (hi ? 8 : 0);
            nb[m * KP + n] = (_Float16)fast_tanh(c[v]);
        }
        __syncthreads();
        p ^= 1;
    }

    // ---- output projection: out[m, o] = curr[m,:] . Wo[o,:] + bo[o] ----
    if (tid < 16 * OUT_SZ) {
        int m = tid / OUT_SZ, o = tid % OUT_SZ;
        float acc = bo[o];
        const _Float16* cf = &cur[p][m * KP];
        for (int k = 0; k < H_SZ; ++k)
            acc += (float)cf[k] * Wo[o * H_SZ + k];
        out[(size_t)(mb + m) * OUT_SZ + o] = acc;
    }
}

extern "C" void kernel_launch(void* const* d_in, const int* in_sizes, int n_in,
                              void* d_out, int out_size, void* d_ws, size_t ws_size,
                              hipStream_t stream) {
    const float* x     = (const float*)d_in[0];
    const int*   order = (const int*)  d_in[1];
    const float* Wi    = (const float*)d_in[2];
    const float* Ws    = (const float*)d_in[3];
    const float* bs    = (const float*)d_in[4];
    const float* Wo    = (const float*)d_in[5];
    const float* bo    = (const float*)d_in[6];
    float* out = (float*)d_out;

    dim3 grid(B_SZ / 16);     // 128 workgroups, one per 16 batch rows
    dim3 block(NW * 32);      // 7 wave32's: one per 16-wide hidden tile
    RecurrentMNIST_scan_kernel<<<grid, block, 0, stream>>>(
        x, order, Wi, Ws, bs, Wo, bo, out);
}
